// CrossKDDenseNet_46265387712941
// MI455X (gfx1250) — matrix operand order repaired and checked by reading, not compiled, not run on hardware
//
#include <hip/hip_runtime.h>
#include <hip/hip_bf16.h>
#include <math.h>

// ---------------------------------------------------------------------------
// Problem constants (match reference)
// ---------------------------------------------------------------------------
#define BROWS  32768
#define DMODEL 688
#define DPAD   704          // 688 padded: multiple of 32 (K) and 64 (N tiles)
#define NHEADS 4
#define DHEAD  172
#define MHID   128
#define LNEPS  1e-5f
#define ATT_SCALE 0.07624928516630233f   // 1/sqrt(172)

typedef __bf16 v16bf __attribute__((ext_vector_type(16)));
typedef __bf16 v8bf  __attribute__((ext_vector_type(8)));
typedef float  v8f   __attribute__((ext_vector_type(8)));
typedef int    v4i   __attribute__((ext_vector_type(4)));

typedef __attribute__((address_space(1))) v4i  gas_v4i;
typedef __attribute__((address_space(3))) v4i  las_v4i;
typedef __attribute__((address_space(3))) char las_char;

// ---------------------------------------------------------------------------
// CDNA5 async global->LDS copy (ASYNCcnt path). Builtin when declared,
// inline-asm mnemonic per CDNA5 ISA otherwise.
// Builtin signature (from hipcc diagnostic): (v4i AS1*, v4i AS3*, Ii, Ii)
// ---------------------------------------------------------------------------
__device__ __forceinline__ void async_copy_b128(const void* g, void* l) {
#if defined(__HIP_DEVICE_COMPILE__)
#if __has_builtin(__builtin_amdgcn_global_load_async_to_lds_b128)
  __builtin_amdgcn_global_load_async_to_lds_b128(
      (gas_v4i*)g, (las_v4i*)l, 0, 0);
#else
  unsigned lofs = (unsigned)(size_t)(las_char*)l;
  unsigned long long ga = (unsigned long long)(size_t)g;
  asm volatile("global_load_async_to_lds_b128 %0, %1, off"
               :: "v"(lofs), "v"(ga) : "memory");
#endif
#endif
}

__device__ __forceinline__ void async_wait0() {
#if defined(__HIP_DEVICE_COMPILE__)
#if __has_builtin(__builtin_amdgcn_s_wait_asynccnt)
  __builtin_amdgcn_s_wait_asynccnt(0);
#else
  asm volatile("s_wait_asynccnt 0x0" ::: "memory");
#endif
#endif
}

// ---------------------------------------------------------------------------
// Weight packing: W [N][K] f32  ->  Wp [Np][Kp] bf16, zero padded
// ---------------------------------------------------------------------------
__global__ __launch_bounds__(256)
void k_pack_w(const float* __restrict__ W, int N, int K, int Np, int Kp,
              __bf16* __restrict__ out) {
  size_t idx = (size_t)blockIdx.x * 256 + threadIdx.x;
  size_t tot = (size_t)Np * Kp;
  if (idx >= tot) return;
  int n = (int)(idx / Kp);
  int k = (int)(idx % Kp);
  float v = (n < N && k < K) ? W[(size_t)n * K + k] : 0.f;
  out[idx] = (__bf16)v;
}

// ---------------------------------------------------------------------------
// LayerNorm (biased var) + pack to bf16 padded row [Dp]
// ---------------------------------------------------------------------------
__global__ __launch_bounds__(256)
void k_ln_pack(const float* __restrict__ X, const float* __restrict__ g,
               const float* __restrict__ b, __bf16* __restrict__ Y) {
  __shared__ float s1[256];
  __shared__ float s2[256];
  const int r = blockIdx.x;
  const int tid = threadIdx.x;
  const float* xr = X + (size_t)r * DMODEL;
  float s = 0.f, q = 0.f;
  for (int k = tid; k < DMODEL; k += 256) {
    float v = xr[k];
    s += v; q += v * v;
  }
  s1[tid] = s; s2[tid] = q;
  __syncthreads();
  for (int w = 128; w > 0; w >>= 1) {
    if (tid < w) { s1[tid] += s1[tid + w]; s2[tid] += s2[tid + w]; }
    __syncthreads();
  }
  const float mean = s1[0] * (1.0f / DMODEL);
  const float var  = s2[0] * (1.0f / DMODEL) - mean * mean;
  const float rstd = rsqrtf(var + LNEPS);
  __bf16* yr = Y + (size_t)r * DPAD;
  for (int k = tid; k < DPAD; k += 256) {
    float v = (k < DMODEL) ? ((xr[k] - mean) * rstd * g[k] + b[k]) : 0.f;
    yr[k] = (__bf16)v;
  }
}

// ---------------------------------------------------------------------------
// WMMA GEMM:  C[B x Nvalid] = A[B x Kp] * Wp[Np x Kp]^T  (+bias, epilogue)
//   block = 128 threads (4 waves), block tile 128(M) x 64(N)
//   wave tile = 32(M) x 64(N): 8 accumulators, K step 32 (bf16)
//   B panel (64 cols x 64 K) double-buffered in LDS, filled with CDNA5
//   async global->LDS b128 copies; compute reads B via ds_load.
// A-fragment (16x32 bf16, contiguous K order): lane&15 = row M, lane>>4 = K-half
//   elem i -> K = i + 16*(lane>>4)                     (two B128 loads)
// B-fragment (32x16 bf16): lane&15 = col N, elem i -> K = i + 16*(lane>>4)
// C/D (16x16 f32): elem r -> M = r + 8*(lane>>4), N = lane&15
// MODE 0: bf16 store (c+bias)            -> outB  (stride outStride)
// MODE 1: f32  store ca*resid + cb*(c+b) -> outF  (resid may alias outF)
// MODE 2: bf16 store gelu_exact(c+bias)  -> outB
// ---------------------------------------------------------------------------
template <int MODE>
__global__ __launch_bounds__(128)
void k_wmma_gemm(const __bf16* __restrict__ A, int Kp,
                 const __bf16* __restrict__ Wp,
                 const float* __restrict__ bias, int Nvalid,
                 float* __restrict__ outF,
                 __bf16* __restrict__ outB, int outStride,
                 const float* __restrict__ resid, int residStride,
                 const float* __restrict__ coef, int ia, int ib) {
  __shared__ __bf16 bbuf[2][64 * 64];   // double-buffered B panel, 16 KB

  const int tid  = threadIdx.x;
  const int lane = tid & 31;
  const int wid  = tid >> 5;
  const int half = lane >> 4;     // K-half selector
  const int l16  = lane & 15;
  const int mBase = blockIdx.x * 128 + wid * 32;
  const int nBase = blockIdx.y * 64;

  v8f acc[8];
#pragma unroll
  for (int t = 0; t < 8; ++t)
#pragma unroll
    for (int i = 0; i < 8; ++i) acc[t][i] = 0.f;

  const __bf16* aRow0 = A + (size_t)(mBase + l16) * Kp + 16 * half;
  const __bf16* aRow1 = aRow0 + (size_t)16 * Kp;

  // async fill of one 64(col) x 64(K) B panel: 512 b128 chunks, 4/thread
  auto fill_panel = [&](int p, int buf) {
#pragma unroll
    for (int i = tid; i < 512; i += 128) {
      const int c = i >> 3;        // col in panel
      const int w = i & 7;         // 16B chunk within 128B row-slice
      const __bf16* g = Wp + (size_t)(nBase + c) * Kp + p * 64 + w * 8;
      __bf16* l = &bbuf[buf][c * 64 + w * 8];
      async_copy_b128(g, l);
    }
  };

  const int npan = Kp >> 6;        // K panels of 64
  fill_panel(0, 0);
  async_wait0();
  __syncthreads();

  const int ldsColOfs = l16 * 64 + 16 * half;   // per-lane B read base

  for (int p = 0; p < npan; ++p) {
    const int cur = p & 1;
    if (p + 1 < npan) fill_panel(p + 1, cur ^ 1);

    const __bf16* bBase = &bbuf[cur][ldsColOfs];
    const int kg = p * 64;
#pragma unroll
    for (int kk = 0; kk < 64; kk += 32) {
      v8bf a0lo = *(const v8bf*)(aRow0 + kg + kk);
      v8bf a0hi = *(const v8bf*)(aRow0 + kg + kk + 8);
      v8bf a1lo = *(const v8bf*)(aRow1 + kg + kk);
      v8bf a1hi = *(const v8bf*)(aRow1 + kg + kk + 8);
      v16bf a0 = __builtin_shufflevector(a0lo, a0hi,
          0, 1, 2, 3, 4, 5, 6, 7, 8, 9, 10, 11, 12, 13, 14, 15);
      v16bf a1 = __builtin_shufflevector(a1lo, a1hi,
          0, 1, 2, 3, 4, 5, 6, 7, 8, 9, 10, 11, 12, 13, 14, 15);
#pragma unroll
      for (int t = 0; t < 4; ++t) {
        v16bf bf = *(const v16bf*)(bBase + t * 16 * 64 + kk);
        acc[t] = __builtin_amdgcn_wmma_f32_16x16x32_bf16(
            false, a0, false, bf, (short)0, acc[t], false, false);
        acc[t + 4] = __builtin_amdgcn_wmma_f32_16x16x32_bf16(
            false, a1, false, bf, (short)0, acc[t + 4], false, false);
      }
    }
    async_wait0();
    __syncthreads();
  }

  float ca = 1.f, cb = 1.f;
  if (MODE == 1) { ca = coef[ia]; cb = coef[ib]; }

#pragma unroll
  for (int t = 0; t < 4; ++t) {
    const int col = nBase + t * 16 + l16;
    if (col >= Nvalid) continue;
    const float bv = bias[col];
#pragma unroll
    for (int fi = 0; fi < 2; ++fi) {
#pragma unroll
      for (int r = 0; r < 8; ++r) {
        const int row = mBase + fi * 16 + r + 8 * half;
        const float c = acc[t + 4 * fi][r] + bv;
        const size_t o = (size_t)row * outStride + col;
        if (MODE == 0) {
          outB[o] = (__bf16)c;
        } else if (MODE == 1) {
          outF[o] = ca * resid[(size_t)row * residStride + col] + cb * c;
        } else {
          // exact GELU: 0.5*x*(1+erf(x/sqrt(2)))
          const float g = 0.5f * c * (1.f + erff(c * 0.70710678118654752f));
          outB[o] = (__bf16)g;
        }
      }
    }
  }
}

// ---------------------------------------------------------------------------
// Per-token cross attention over heads (scores are 4x4 per token)
//   out_vis = softmax(q_ir . k_vis * s) @ v_vis
//   out_ir  = softmax(q_vis . k_ir * s) @ v_ir
// Writes bf16, padded to DPAD columns for the Wo WMMA GEMMs.
// ---------------------------------------------------------------------------
__global__ __launch_bounds__(128)
void k_cross_attn(const __bf16* __restrict__ qv, const __bf16* __restrict__ kv,
                  const __bf16* __restrict__ vv, const __bf16* __restrict__ qi,
                  const __bf16* __restrict__ ki, const __bf16* __restrict__ vi,
                  __bf16* __restrict__ attnV, __bf16* __restrict__ attnI) {
  __shared__ float sh[6 * DMODEL + 64];
  float* Qv = sh;
  float* Kv = sh + DMODEL;
  float* Vv = sh + 2 * DMODEL;
  float* Qi = sh + 3 * DMODEL;
  float* Ki = sh + 4 * DMODEL;
  float* Vi = sh + 5 * DMODEL;
  float* sc  = sh + 6 * DMODEL;   // 32 scores: [0..15] vis-path, [16..31] ir-path
  float* att = sc + 32;

  const int r = blockIdx.x;
  const int tid = threadIdx.x;
  const size_t base = (size_t)r * DMODEL;
  for (int k = tid; k < DMODEL; k += 128) {
    Qv[k] = (float)qv[base + k]; Kv[k] = (float)kv[base + k]; Vv[k] = (float)vv[base + k];
    Qi[k] = (float)qi[base + k]; Ki[k] = (float)ki[base + k]; Vi[k] = (float)vi[base + k];
  }
  __syncthreads();

  if (tid < 32) {
    const int which = tid >> 4;            // 0 = vis-path, 1 = ir-path
    const int h = (tid >> 2) & 3, g = tid & 3;
    const float* Q = which ? Qv : Qi;      // vis-path uses q_ir
    const float* K = which ? Ki : Kv;      // vis-path uses k_vis
    float sum = 0.f;
    for (int d = 0; d < DHEAD; ++d) sum += Q[h * DHEAD + d] * K[g * DHEAD + d];
    sc[tid] = sum * ATT_SCALE;
  }
  __syncthreads();

  if (tid < 8) {
    const int basei = (tid >> 2) * 16 + (tid & 3) * 4;
    const float m = fmaxf(fmaxf(sc[basei], sc[basei + 1]),
                          fmaxf(sc[basei + 2], sc[basei + 3]));
    const float e0 = expf(sc[basei] - m),     e1 = expf(sc[basei + 1] - m);
    const float e2 = expf(sc[basei + 2] - m), e3 = expf(sc[basei + 3] - m);
    const float inv = 1.f / (e0 + e1 + e2 + e3);
    att[basei] = e0 * inv; att[basei + 1] = e1 * inv;
    att[basei + 2] = e2 * inv; att[basei + 3] = e3 * inv;
  }
  __syncthreads();

  for (int j = tid; j < DPAD; j += 128) {
    float ov = 0.f, oi = 0.f;
    if (j < DMODEL) {
      const int h = j / DHEAD, dd = j - h * DHEAD;
#pragma unroll
      for (int g = 0; g < 4; ++g) {
        ov += att[h * 4 + g]      * Vv[g * DHEAD + dd];
        oi += att[16 + h * 4 + g] * Vi[g * DHEAD + dd];
      }
    }
    attnV[(size_t)r * DPAD + j] = (__bf16)ov;
    attnI[(size_t)r * DPAD + j] = (__bf16)oi;
  }
}

// ---------------------------------------------------------------------------
// Host-side orchestration
// ---------------------------------------------------------------------------
static inline void launch_pack(const float* W, int N, int K, int Np, int Kp,
                               __bf16* out, hipStream_t s) {
  size_t tot = (size_t)Np * Kp;
  int blocks = (int)((tot + 255) / 256);
  k_pack_w<<<blocks, 256, 0, s>>>(W, N, K, Np, Kp, out);
}

extern "C" void kernel_launch(void* const* d_in, const int* in_sizes, int n_in,
                              void* d_out, int out_size, void* d_ws, size_t ws_size,
                              hipStream_t stream) {
  (void)in_sizes; (void)n_in; (void)out_size; (void)ws_size;
  // Inputs per setup_inputs() order
  const float* x    = (const float*)d_in[0];
  const float* x2   = (const float*)d_in[1];
  const float* Wq_v = (const float*)d_in[2];  const float* bq_v = (const float*)d_in[3];
  const float* Wk_v = (const float*)d_in[4];  const float* bk_v = (const float*)d_in[5];
  const float* Wv_v = (const float*)d_in[6];  const float* bv_v = (const float*)d_in[7];
  const float* Wq_i = (const float*)d_in[8];  const float* bq_i = (const float*)d_in[9];
  const float* Wk_i = (const float*)d_in[10]; const float* bk_i = (const float*)d_in[11];
  const float* Wv_i = (const float*)d_in[12]; const float* bv_i = (const float*)d_in[13];
  const float* Wo_v = (const float*)d_in[14]; const float* bo_v = (const float*)d_in[15];
  const float* Wo_i = (const float*)d_in[16]; const float* bo_i = (const float*)d_in[17];
  const float* ln1_g = (const float*)d_in[18]; const float* ln1_b = (const float*)d_in[19];
  const float* ln2_g = (const float*)d_in[20]; const float* ln2_b = (const float*)d_in[21];
  const float* ln3_g = (const float*)d_in[22]; const float* ln3_b = (const float*)d_in[23];
  const float* ln4_g = (const float*)d_in[24]; const float* ln4_b = (const float*)d_in[25];
  const float* m1v_W = (const float*)d_in[26]; const float* m1v_b = (const float*)d_in[27];
  const float* m2v_W = (const float*)d_in[28]; const float* m2v_b = (const float*)d_in[29];
  const float* m1i_W = (const float*)d_in[30]; const float* m1i_b = (const float*)d_in[31];
  const float* m2i_W = (const float*)d_in[32]; const float* m2i_b = (const float*)d_in[33];
  const float* coef  = (const float*)d_in[34];

  float* outVis = (float*)d_out;
  float* outIr  = (float*)d_out + (size_t)BROWS * DMODEL;

  // ---- workspace carve (256B aligned bump allocator) ----
  char* p = (char*)d_ws;
  auto alloc = [&](size_t bytes) -> void* {
    void* r = (void*)p;
    p += (bytes + 255) & ~(size_t)255;
    return r;
  };
  const size_t szWD  = (size_t)DPAD * DPAD * sizeof(__bf16);   // 704x704 bf16
  const size_t szW1  = (size_t)MHID * DPAD * sizeof(__bf16);   // 128x704
  const size_t szW2  = (size_t)DPAD * MHID * sizeof(__bf16);   // 704x128
  const size_t szAp  = (size_t)BROWS * DPAD * sizeof(__bf16);  // padded act
  const size_t szAq  = (size_t)BROWS * DMODEL * sizeof(__bf16);// qkv act

  __bf16* wp_qv = (__bf16*)alloc(szWD); __bf16* wp_kv = (__bf16*)alloc(szWD);
  __bf16* wp_vv = (__bf16*)alloc(szWD); __bf16* wp_qi = (__bf16*)alloc(szWD);
  __bf16* wp_ki = (__bf16*)alloc(szWD); __bf16* wp_vi = (__bf16*)alloc(szWD);
  __bf16* wp_ov = (__bf16*)alloc(szWD); __bf16* wp_oi = (__bf16*)alloc(szWD);
  __bf16* wp_m1v = (__bf16*)alloc(szW1); __bf16* wp_m1i = (__bf16*)alloc(szW1);
  __bf16* wp_m2v = (__bf16*)alloc(szW2); __bf16* wp_m2i = (__bf16*)alloc(szW2);
  __bf16* rgb  = (__bf16*)alloc(szAp);   // ln1 out; reused for ln3 out
  __bf16* irb  = (__bf16*)alloc(szAp);   // ln2 out; reused for ln4 out
  __bf16* qv = (__bf16*)alloc(szAq); __bf16* kv = (__bf16*)alloc(szAq);
  __bf16* vv = (__bf16*)alloc(szAq); __bf16* qi = (__bf16*)alloc(szAq);
  __bf16* ki = (__bf16*)alloc(szAq); __bf16* vi = (__bf16*)alloc(szAq);
  __bf16* attnV = (__bf16*)alloc(szAp);
  __bf16* attnI = (__bf16*)alloc(szAp);
  __bf16* hv = (__bf16*)qv;   // reuse qkv region once attention is done
  __bf16* hi = (__bf16*)kv;

  // ---- 1. pack weights to padded bf16 ----
  launch_pack(Wq_v, DMODEL, DMODEL, DPAD, DPAD, wp_qv, stream);
  launch_pack(Wk_v, DMODEL, DMODEL, DPAD, DPAD, wp_kv, stream);
  launch_pack(Wv_v, DMODEL, DMODEL, DPAD, DPAD, wp_vv, stream);
  launch_pack(Wq_i, DMODEL, DMODEL, DPAD, DPAD, wp_qi, stream);
  launch_pack(Wk_i, DMODEL, DMODEL, DPAD, DPAD, wp_ki, stream);
  launch_pack(Wv_i, DMODEL, DMODEL, DPAD, DPAD, wp_vi, stream);
  launch_pack(Wo_v, DMODEL, DMODEL, DPAD, DPAD, wp_ov, stream);
  launch_pack(Wo_i, DMODEL, DMODEL, DPAD, DPAD, wp_oi, stream);
  launch_pack(m1v_W, MHID, DMODEL, MHID, DPAD, wp_m1v, stream);
  launch_pack(m1i_W, MHID, DMODEL, MHID, DPAD, wp_m1i, stream);
  launch_pack(m2v_W, DMODEL, MHID, DPAD, MHID, wp_m2v, stream);
  launch_pack(m2i_W, DMODEL, MHID, DPAD, MHID, wp_m2i, stream);

  // ---- 2. LN1/LN2 + bf16 pack ----
  k_ln_pack<<<BROWS, 256, 0, stream>>>(x,  ln1_g, ln1_b, rgb);
  k_ln_pack<<<BROWS, 256, 0, stream>>>(x2, ln2_g, ln2_b, irb);

  // ---- 3. QKV projections (6 GEMMs, bf16 out) ----
  dim3 gD(BROWS / 128, DPAD / 64);   // (256, 11)
  dim3 blk(128);
  k_wmma_gemm<0><<<gD, blk, 0, stream>>>(rgb, DPAD, wp_qv, bq_v, DMODEL,
      nullptr, qv, DMODEL, nullptr, 0, nullptr, 0, 0);
  k_wmma_gemm<0><<<gD, blk, 0, stream>>>(rgb, DPAD, wp_kv, bk_v, DMODEL,
      nullptr, kv, DMODEL, nullptr, 0, nullptr, 0, 0);
  k_wmma_gemm<0><<<gD, blk, 0, stream>>>(rgb, DPAD, wp_vv, bv_v, DMODEL,
      nullptr, vv, DMODEL, nullptr, 0, nullptr, 0, 0);
  k_wmma_gemm<0><<<gD, blk, 0, stream>>>(irb, DPAD, wp_qi, bq_i, DMODEL,
      nullptr, qi, DMODEL, nullptr, 0, nullptr, 0, 0);
  k_wmma_gemm<0><<<gD, blk, 0, stream>>>(irb, DPAD, wp_ki, bk_i, DMODEL,
      nullptr, ki, DMODEL, nullptr, 0, nullptr, 0, 0);
  k_wmma_gemm<0><<<gD, blk, 0, stream>>>(irb, DPAD, wp_vi, bv_i, DMODEL,
      nullptr, vi, DMODEL, nullptr, 0, nullptr, 0, 0);

  // ---- 4. per-token cross attention (both directions) ----
  k_cross_attn<<<BROWS, 128, 0, stream>>>(qv, kv, vv, qi, ki, vi, attnV, attnI);

  // ---- 5. output projections + residual: out = c0*x + c1*(attn@Wo^T + bo) ----
  k_wmma_gemm<1><<<gD, blk, 0, stream>>>(attnV, DPAD, wp_ov, bo_v, DMODEL,
      outVis, nullptr, DMODEL, x,  DMODEL, coef, 0, 1);
  k_wmma_gemm<1><<<gD, blk, 0, stream>>>(attnI, DPAD, wp_oi, bo_i, DMODEL,
      outIr,  nullptr, DMODEL, x2, DMODEL, coef, 2, 3);

  // ---- 6. LN3/LN4 on residual streams (reuse rgb/irb buffers) ----
  k_ln_pack<<<BROWS, 256, 0, stream>>>(outVis, ln3_g, ln3_b, rgb);
  k_ln_pack<<<BROWS, 256, 0, stream>>>(outIr,  ln4_g, ln4_b, irb);

  // ---- 7. MLP up-proj + exact GELU (bf16 out) ----
  dim3 gM(BROWS / 128, MHID / 64);   // (256, 2)
  k_wmma_gemm<2><<<gM, blk, 0, stream>>>(rgb, DPAD, wp_m1v, m1v_b, MHID,
      nullptr, hv, MHID, nullptr, 0, nullptr, 0, 0);
  k_wmma_gemm<2><<<gM, blk, 0, stream>>>(irb, DPAD, wp_m1i, m1i_b, MHID,
      nullptr, hi, MHID, nullptr, 0, nullptr, 0, 0);

  // ---- 8. MLP down-proj + final combine (in-place on d_out) ----
  k_wmma_gemm<1><<<gD, blk, 0, stream>>>(hv, MHID, wp_m2v, m2v_b, DMODEL,
      outVis, nullptr, DMODEL, outVis, DMODEL, coef, 4, 5);
  k_wmma_gemm<1><<<gD, blk, 0, stream>>>(hi, MHID, wp_m2i, m2i_b, DMODEL,
      outIr,  nullptr, DMODEL, outIr,  DMODEL, coef, 6, 7);
}
